// NetMamba_20169166422722
// MI455X (gfx1250) — matrix operand order repaired and checked by reading, not compiled, not run on hardware
//
#include <hip/hip_runtime.h>
#include <hip/hip_bf16.h>
#include <math.h>

// ---------------------------------------------------------------------------
// NetMamba forward for MI455X (gfx1250), wave32.
//  - ALL matmuls (in_proj / x_proj / out_proj / head) on v_wmma_f32_16x16x32_bf16
//    with f32 accumulation; activations travel as bf16 (halves HBM bytes).
//  - Selective scan: 1 thread per (batch,channel); 16-state kept in VGPRs;
//    per-step B/C (32 floats) broadcast via LDS; 401 serial steps.
// ---------------------------------------------------------------------------

typedef __attribute__((ext_vector_type(16))) __bf16 v16bf;
typedef __attribute__((ext_vector_type(8)))  __bf16 v8bf;
typedef __attribute__((ext_vector_type(8)))  float  v8f;

#define BATCH   128
#define SEQL    401
#define DM      192
#define DI      384      // d_inner
#define DS      16       // d_state
#define DTR     12       // dt_rank
#define DBLW    44       // DTR + 2*DS (logical)
#define DBLP    48       // padded x_proj output width (multiple of 16)
#define NDEPTH  4
#define MROWS   (BATCH * SEQL)   // 51328 (= 1604 * 32)
#define NCLS    1000
#define NCLSP   1024             // padded head rows
#define REPS    1e-5f

// ---------------- elementwise f32 -> bf16 (weight pre-conversion) ----------
__global__ void k_f32_to_bf16(const float* __restrict__ s, __bf16* __restrict__ d, int n) {
  int i = blockIdx.x * blockDim.x + threadIdx.x;
  if (i < n) d[i] = (__bf16)s[i];
}

// ---------------- x_proj weight: (4,44,384) f32 -> (4,48,384) bf16 padded --
__global__ void k_pad_xpw(const float* __restrict__ xw, __bf16* __restrict__ dst) {
  int t = blockIdx.x * blockDim.x + threadIdx.x;
  if (t >= NDEPTH * DBLP * DI) return;
  int k = t % DI;
  int e = (t / DI) % DBLP;
  int layer = t / (DI * DBLP);
  dst[t] = (e < DBLW) ? (__bf16)xw[((size_t)layer * DBLW + e) * DI + k] : (__bf16)0.f;
}

// ---------------- head weight: (1000,192) f32 -> (1024,192) bf16 padded ----
__global__ void k_pad_headw(const float* __restrict__ hw, __bf16* __restrict__ dst) {
  int t = blockIdx.x * blockDim.x + threadIdx.x;
  if (t >= NCLSP * DM) return;
  int k = t % DM;
  int c = t / DM;
  dst[t] = (c < NCLS) ? (__bf16)hw[(size_t)c * DM + k] : (__bf16)0.f;
}

// ---------------- patch embed + pos embed + cls, zero residual -------------
__global__ void k_patch(const float* __restrict__ imgs, const float* __restrict__ pw,
                        const float* __restrict__ pb, const float* __restrict__ pos,
                        const float* __restrict__ cls,
                        float* __restrict__ H, float* __restrict__ RESID) {
  int t = blockIdx.x * blockDim.x + threadIdx.x;
  if (t >= MROWS * DM) return;
  int d   = t % DM;
  int row = t / DM;
  int l   = row % SEQL;
  int b   = row / SEQL;
  float acc;
  if (l < SEQL - 1) {
    const float* px = imgs + (size_t)b * 1600 + l * 4;
    const float* w  = pw + d * 4;
    acc = pb[d] + px[0]*w[0] + px[1]*w[1] + px[2]*w[2] + px[3]*w[3] + pos[l * DM + d];
  } else {
    acc = cls[d] + pos[(SEQL - 1) * DM + d];
  }
  H[t]     = acc;
  RESID[t] = 0.f;
}

// ---------------- residual += h ; normed = rmsnorm(residual) (bf16) --------
__global__ void k_addnorm(const float* __restrict__ H, float* __restrict__ RESID,
                          const float* __restrict__ w, __bf16* __restrict__ NOUT) {
  const int wave = threadIdx.x >> 5, lane = threadIdx.x & 31;
  const int row  = blockIdx.x * 8 + wave;           // one wave32 per 192-elem row
  if (row >= MROWS) return;
  const float* hr = H + (size_t)row * DM;
  float*       rr = RESID + (size_t)row * DM;
  float s[6], acc = 0.f;
  #pragma unroll
  for (int j = 0; j < 6; ++j) {
    int idx = lane + j * 32;
    float v = hr[idx] + rr[idx];
    rr[idx] = v;
    s[j] = v;
    acc += v * v;
  }
  #pragma unroll
  for (int off = 16; off > 0; off >>= 1) acc += __shfl_xor(acc, off, 32);
  float sc = rsqrtf(acc * (1.f / DM) + REPS);
  #pragma unroll
  for (int j = 0; j < 6; ++j) {
    int idx = lane + j * 32;
    NOUT[(size_t)row * DM + idx] = (__bf16)(s[j] * sc * w[idx]);
  }
}

// ---------------- bf16 WMMA GEMM: out[M,*] = A[M,K] * W[*,K]^T -------------
// Wave tile 16x16, K-step 32.  Fragment layouts follow CDNA5 ISA 7.12.2:
//   A 16x32 bf16: lane<16 row=lane holds K{0..7,16..23}; lane>=16 K{8..15,24..31}
//   B 32x16 bf16: lane<16 col=lane holds K{0..7,16..23}; lane>=16 K{8..15,24..31}
//   C/D v8f:      lane<16 -> M=r, lane>=16 -> M=r+8, N=lane&15
// Block tile: 32 x (16 * WN), WN = (blockDim.x/32)/2; M-grid = M/32 exact.
// Store guarded by nBound (lets us pad N); optional fused bias (f32 out).
template <bool OUT_BF16>
__global__ void k_gemm_wmma(const __bf16* __restrict__ A, const __bf16* __restrict__ W,
                            void* __restrict__ out, int K, int ldo, int nBound,
                            const float* __restrict__ bias) {
  const int lane = threadIdx.x & 31;
  const int wave = threadIdx.x >> 5;
  const int WN   = (blockDim.x >> 5) >> 1;           // wave columns per block
  const int m0 = blockIdx.x * 32 + (wave & 1) * 16;
  const int n0 = blockIdx.y * (16 * WN) + (wave >> 1) * 16;
  const int hi = lane >> 4;
  const int mr = m0 + (lane & 15);
  const int nr = n0 + (lane & 15);
  const __bf16* arow = A + (size_t)mr * K;
  const __bf16* wrow = W + (size_t)nr * K;
  v8f acc = {};
  for (int k0 = 0; k0 < K; k0 += 32) {
    __builtin_prefetch(wrow + k0 + 64, 0, 1);        // global_prefetch_b8 on weight stream
    const int ka = k0 + hi * 8;
    v8bf alo = *reinterpret_cast<const v8bf*>(arow + ka);
    v8bf ahi = *reinterpret_cast<const v8bf*>(arow + ka + 16);
    v16bf av = __builtin_shufflevector(alo, ahi, 0,1,2,3,4,5,6,7,8,9,10,11,12,13,14,15);
    v8bf blo = *reinterpret_cast<const v8bf*>(wrow + ka);
    v8bf bhi = *reinterpret_cast<const v8bf*>(wrow + ka + 16);
    v16bf bv = __builtin_shufflevector(blo, bhi, 0,1,2,3,4,5,6,7,8,9,10,11,12,13,14,15);
    acc = __builtin_amdgcn_wmma_f32_16x16x32_bf16(false, av, false, bv,
                                                  (short)0, acc, false, false);
  }
  const int nc = n0 + (lane & 15);
  if (nc >= nBound) return;
  if constexpr (OUT_BF16) {
    __bf16* o = (__bf16*)out;
    #pragma unroll
    for (int r = 0; r < 8; ++r) o[(size_t)(m0 + hi * 8 + r) * ldo + nc] = (__bf16)acc[r];
  } else {
    float* o = (float*)out;
    const float bb = bias ? bias[nc] : 0.f;
    #pragma unroll
    for (int r = 0; r < 8; ++r) o[(size_t)(m0 + hi * 8 + r) * ldo + nc] = acc[r] + bb;
  }
}

// ---------------- depthwise causal conv1d (k=4) + bias + SiLU --------------
__global__ void k_conv_silu(const __bf16* __restrict__ XZ, const float* __restrict__ cw,
                            const float* __restrict__ cb, __bf16* __restrict__ XT) {
  int t = blockIdx.x * blockDim.x + threadIdx.x;
  if (t >= MROWS * DI) return;
  int d    = t % DI;
  int row  = t / DI;
  int l    = row % SEQL;
  int base = row - l;                                 // b*SEQL
  float acc = cb[d];
  const float* w = cw + d * 4;
  #pragma unroll
  for (int j = 0; j < 4; ++j) {
    int ll = l - 3 + j;
    if (ll >= 0) acc += w[j] * (float)XZ[(size_t)(base + ll) * (2 * DI) + d];
  }
  XT[t] = (__bf16)(acc / (1.f + __expf(-acc)));
}

// ---------------- dt = softplus(dt_low * dtw^T + dtb) ----------------------
// DBL is the padded (M, 48) x_proj output: [0..11]=dt_low, [12..27]=B, [28..43]=C
__global__ void k_dtproj(const float* __restrict__ DBL, const float* __restrict__ dtw,
                         const float* __restrict__ dtb, __bf16* __restrict__ DT) {
  int t = blockIdx.x * blockDim.x + threadIdx.x;
  if (t >= MROWS * DI) return;
  int d   = t % DI;
  int row = t / DI;
  const float* dl = DBL + (size_t)row * DBLP;
  const float* w  = dtw + d * DTR;
  float acc = dtb[d];
  #pragma unroll
  for (int r = 0; r < DTR; ++r) acc += dl[r] * w[r];
  float sp = (acc > 20.f) ? acc : __logf(1.f + __expf(acc));
  DT[t] = (__bf16)sp;
}

// ---------------- selective scan + D-skip + silu(z) gate -------------------
// One block per batch, one thread per channel; 16-state in VGPRs; per-step
// B/C scalars broadcast via LDS.
__global__ void k_scan(const __bf16* __restrict__ DT, const __bf16* __restrict__ XT,
                       const float* __restrict__ DBL, const __bf16* __restrict__ XZ,
                       const float* __restrict__ A_log, const float* __restrict__ Dsk,
                       __bf16* __restrict__ Y) {
  const int b = blockIdx.x;
  const int d = threadIdx.x;                          // 0..383
  __shared__ float sB[DS], sC[DS];
  float A[DS], h[DS];
  #pragma unroll
  for (int n = 0; n < DS; ++n) {
    A[n] = -__expf(A_log[(size_t)d * DS + n]);
    h[n] = 0.f;
  }
  const float dsk = Dsk[d];
  for (int l = 0; l < SEQL; ++l) {
    const size_t row = (size_t)b * SEQL + l;
    if (d < 32) {
      float v = DBL[row * DBLP + DTR + d];
      if (d < DS) sB[d] = v; else sC[d - DS] = v;
    }
    __syncthreads();
    float dt = (float)DT[row * DI + d];
    float x  = (float)XT[row * DI + d];
    float z  = (float)XZ[row * (2 * DI) + DI + d];
    float dx = dt * x;
    float y = 0.f;
    #pragma unroll
    for (int n = 0; n < DS; ++n) {
      h[n] = __expf(dt * A[n]) * h[n] + dx * sB[n];
      y += h[n] * sC[n];
    }
    y += x * dsk;
    y *= z / (1.f + __expf(-z));
    Y[row * DI + d] = (__bf16)y;
    __syncthreads();
  }
}

// ---------------- final add + RMSNorm on cls token only (bf16 out) ---------
__global__ void k_final_norm(const float* __restrict__ H, const float* __restrict__ RESID,
                             const float* __restrict__ w, __bf16* __restrict__ CLSN) {
  const int wave = threadIdx.x >> 5, lane = threadIdx.x & 31;
  const int b = blockIdx.x * 8 + wave;
  if (b >= BATCH) return;
  const size_t row = (size_t)b * SEQL + (SEQL - 1);
  float s[6], acc = 0.f;
  #pragma unroll
  for (int j = 0; j < 6; ++j) {
    int idx = lane + j * 32;
    float v = H[row * DM + idx] + RESID[row * DM + idx];
    s[j] = v;
    acc += v * v;
  }
  #pragma unroll
  for (int off = 16; off > 0; off >>= 1) acc += __shfl_xor(acc, off, 32);
  float sc = rsqrtf(acc * (1.f / DM) + REPS);
  #pragma unroll
  for (int j = 0; j < 6; ++j) {
    int idx = lane + j * 32;
    CLSN[b * DM + idx] = (__bf16)(s[j] * sc * w[idx]);
  }
}

// ---------------------------------------------------------------------------
extern "C" void kernel_launch(void* const* d_in, const int* in_sizes, int n_in,
                              void* d_out, int out_size, void* d_ws, size_t ws_size,
                              hipStream_t stream) {
  (void)in_sizes; (void)n_in; (void)out_size; (void)ws_size;
  const float* imgs      = (const float*)d_in[0];
  const float* patch_w   = (const float*)d_in[1];
  const float* patch_b   = (const float*)d_in[2];
  const float* pos_embed = (const float*)d_in[3];
  const float* cls_token = (const float*)d_in[4];
  const float* in_proj_w = (const float*)d_in[5];
  const float* conv_w    = (const float*)d_in[6];
  const float* conv_b    = (const float*)d_in[7];
  const float* x_proj_w  = (const float*)d_in[8];
  const float* dt_proj_w = (const float*)d_in[9];
  const float* dt_proj_b = (const float*)d_in[10];
  const float* A_log     = (const float*)d_in[11];
  const float* D_skip    = (const float*)d_in[12];
  const float* out_proj_w= (const float*)d_in[13];
  const float* norm_w    = (const float*)d_in[14];
  const float* norm_f_w  = (const float*)d_in[15];
  const float* head_w    = (const float*)d_in[16];
  const float* head_b    = (const float*)d_in[17];

  // ---- workspace carve-out (~308 MB), 256B aligned ----
  size_t off = 0;
  auto carve = [&](size_t bytes) -> void* {
    void* p = (char*)d_ws + off;
    off += (bytes + 255) & ~(size_t)255;
    return p;
  };
  float*  RESID  = (float*) carve((size_t)MROWS * DM * 4);
  float*  H      = (float*) carve((size_t)MROWS * DM * 4);
  __bf16* NORMED = (__bf16*)carve((size_t)MROWS * DM * 2);
  __bf16* XZ     = (__bf16*)carve((size_t)MROWS * 2 * DI * 2);
  __bf16* XT     = (__bf16*)carve((size_t)MROWS * DI * 2);
  __bf16* DT     = (__bf16*)carve((size_t)MROWS * DI * 2);
  float*  DBL    = (float*) carve((size_t)MROWS * DBLP * 4);
  __bf16* Y      = (__bf16*)carve((size_t)MROWS * DI * 2);
  __bf16* WIN    = (__bf16*)carve((size_t)NDEPTH * 2 * DI * DM * 2);
  __bf16* WOUT   = (__bf16*)carve((size_t)NDEPTH * DM * DI * 2);
  __bf16* XPW    = (__bf16*)carve((size_t)NDEPTH * DBLP * DI * 2);
  __bf16* HWB    = (__bf16*)carve((size_t)NCLSP * DM * 2);
  __bf16* CLSN   = (__bf16*)carve((size_t)BATCH * DM * 2);

  // ---- weight pre-conversion / padding to bf16 (tiny; L2-resident) ----
  {
    int n1 = NDEPTH * 2 * DI * DM;                   // 589824
    int n2 = NDEPTH * DM * DI;                       // 294912
    int n3 = NDEPTH * DBLP * DI;                     // 73728
    int n4 = NCLSP * DM;                             // 196608
    k_f32_to_bf16<<<(n1 + 255) / 256, 256, 0, stream>>>(in_proj_w,  WIN,  n1);
    k_f32_to_bf16<<<(n2 + 255) / 256, 256, 0, stream>>>(out_proj_w, WOUT, n2);
    k_pad_xpw   <<<(n3 + 255) / 256, 256, 0, stream>>>(x_proj_w,  XPW);
    k_pad_headw <<<(n4 + 255) / 256, 256, 0, stream>>>(head_w,    HWB);
  }

  // ---- patch embed (writes H = x0, zeros RESID) ----
  k_patch<<<(MROWS * DM + 255) / 256, 256, 0, stream>>>(
      imgs, patch_w, patch_b, pos_embed, cls_token, H, RESID);

  // ---- 4 mamba layers ----
  for (int layer = 0; layer < NDEPTH; ++layer) {
    k_addnorm<<<MROWS / 8, 256, 0, stream>>>(H, RESID, norm_w + layer * DM, NORMED);

    // in_proj: XZ[51328,768] = NORMED[51328,192] x WIN[768,192]^T
    k_gemm_wmma<true><<<dim3(MROWS / 32, (2 * DI) / 64), 256, 0, stream>>>(
        NORMED, WIN + (size_t)layer * 2 * DI * DM, XZ, DM, 2 * DI, 2 * DI, nullptr);

    k_conv_silu<<<(MROWS * DI + 255) / 256, 256, 0, stream>>>(
        XZ, conv_w + (size_t)layer * DI * 4, conv_b + layer * DI, XT);

    // x_proj: DBL[51328,48] = XT[51328,384] x XPW[48,384]^T  (192 thr -> 32x48 tile)
    k_gemm_wmma<false><<<dim3(MROWS / 32, 1), 192, 0, stream>>>(
        XT, XPW + (size_t)layer * DBLP * DI, DBL, DI, DBLP, DBLP, nullptr);

    k_dtproj<<<(MROWS * DI + 255) / 256, 256, 0, stream>>>(
        DBL, dt_proj_w + (size_t)layer * DI * DTR, dt_proj_b + layer * DI, DT);

    k_scan<<<BATCH, DI, 0, stream>>>(
        DT, XT, DBL, XZ, A_log + (size_t)layer * DI * DS, D_skip + layer * DI, Y);

    // out_proj: H[51328,192] = Y[51328,384] x WOUT[192,384]^T
    k_gemm_wmma<false><<<dim3(MROWS / 32, DM / 64), 256, 0, stream>>>(
        Y, WOUT + (size_t)layer * DM * DI, H, DI, DM, DM, nullptr);
  }

  // ---- final norm on cls token + WMMA head (bias fused, N guarded @1000) ----
  k_final_norm<<<BATCH / 8, 256, 0, stream>>>(H, RESID, norm_f_w, CLSN);
  k_gemm_wmma<false><<<dim3(BATCH / 32, NCLSP / 64), 256, 0, stream>>>(
      CLSN, HWB, d_out, DM, NCLS, NCLS, head_b);
}
